// AspEntQuaNet_82678120448416
// MI455X (gfx1250) — matrix operand order repaired and checked by reading, not compiled, not run on hardware
//
#include <hip/hip_runtime.h>
#include <stdint.h>

// ---------------------------------------------------------------------------
// AspEntQuaNet on MI455X (gfx1250): bf16 WMMA for the two big input-projection
// GEMMs and for the serial LSTM recurrence; fp32 VALU for the tiny MLP head.
// X is pre-packed to bf16 A-fragment order so the GEMM hot loop has zero
// conversion VALU; all fragment loads of a k-step are batched ahead of the
// WMMA chain so load latency overlaps the matrix pipe.
// ---------------------------------------------------------------------------

typedef __attribute__((ext_vector_type(16))) __bf16       v16bf;
typedef __attribute__((ext_vector_type(8)))  float        v8f;
typedef __attribute__((ext_vector_type(4)))  unsigned int v4u;

#define B_   256
#define T_   500
#define V_   768
#define U_   256
#define G_   1024   // 4*U
#define H1_  512
#define H2_  256
#define FEAT_ 534
#define MT_  (B_ * T_ / 16)   // 8000 m-tiles (m' = t*B + b ordering)
#define KTX_ (V_ / 32)        // 24 k-tiles for the projection
#define KTH_ (U_ / 32)        // 8 k-tiles for the recurrence
#define NT_  (G_ / 16)        // 64 n-tiles

__device__ __forceinline__ uint16_t f2bf_u(float f) {
  union { float f; uint32_t u; } v; v.f = f;
  uint32_t r = v.u + 0x7FFFu + ((v.u >> 16) & 1u);   // round-to-nearest-even
  return (uint16_t)(r >> 16);
}
__device__ __forceinline__ float bf2f(uint16_t b) {
  union { uint32_t u; float f; } v; v.u = ((uint32_t)b) << 16;
  return v.f;
}
__device__ __forceinline__ float sigmoidf(float x) {
  return 1.0f / (1.0f + __expf(-x));
}

union ABFrag { v16bf v; uint16_t u[16]; v4u q[2]; };  // 32 bytes / lane
union CQ     { v4u q; uint16_t u[8]; };               // 16 bytes / lane

// ---------------------------------------------------------------------------
// Pack a row-major fp32 [K, N=1024] weight into bf16 WMMA B-fragment order.
// B tile = 32(K) x 16(N): lane l holds col n = l&15, k = ((e<8)?0:16) + (l>>4)*8 + (e&7).
// Tile (kt, nt) contiguous: 512 bf16, per-lane chunk of 16 elems.
// ---------------------------------------------------------------------------
__global__ void pack_b_kernel(const float* __restrict__ src,
                              uint16_t* __restrict__ dst,
                              int total, int N) {
  int idx = blockIdx.x * blockDim.x + threadIdx.x;
  if (idx >= total) return;
  int tile = idx >> 9;          // /512
  int rem  = idx & 511;
  int lane = rem >> 4;
  int e    = rem & 15;
  int kt   = tile >> 6;         // /64 (NT = 64)
  int nt   = tile & 63;
  int k = kt * 32 + ((e < 8) ? 0 : 16) + ((lane >> 4) << 3) + (e & 7);
  int n = nt * 16 + (lane & 15);
  dst[idx] = f2bf_u(src[(size_t)k * N + n]);
}

// ---------------------------------------------------------------------------
// Pack X [B,T,V] fp32 into bf16 WMMA A-fragment order over m' = t*B + b rows.
// A tile = 16(M) x 32(K): lane l holds row m = l&15,
// k = ((e<8)?0:16) + (l>>4)*8 + (e&7)  (ISA dense 16-bit A layout).
// Tile (mt, kt) contiguous: 512 bf16, per-lane chunk of 16 elems.
// ---------------------------------------------------------------------------
__global__ void pack_a_kernel(const float* __restrict__ X,
                              uint16_t* __restrict__ dst,
                              int total) {
  int idx = blockIdx.x * blockDim.x + threadIdx.x;
  if (idx >= total) return;
  int tile = idx >> 9;
  int rem  = idx & 511;
  int lane = rem >> 4;
  int e    = rem & 15;
  int mt   = tile / KTX_;
  int kt   = tile - mt * KTX_;
  int k_local = ((e < 8) ? 0 : 16) + ((lane >> 4) << 3) + (e & 7);
  int mp = mt * 16 + (lane & 15);     // m' = t*256 + b
  int t  = mp >> 8;
  int b  = mp & 255;
  dst[idx] = f2bf_u(X[(size_t)(b * T_ + t) * V_ + kt * 32 + k_local]);
}

// ---------------------------------------------------------------------------
// Projection GEMM: XZ[m', g] = X_row(m') . Wx[:, g] + bias[g].
// M = 128000, N = 1024, K = 768.  Workgroup tile 128x128, 8 waves, each wave
// 32x64 (2x4 WMMA tiles).  A and B both pre-packed bf16; per k-step all six
// fragments are loaded before the 8-WMMA chain so waits are partial.
// Output stored bf16 in C-fragment-packed tile order.
// ---------------------------------------------------------------------------
__global__ __launch_bounds__(256)
void proj_gemm_kernel(const uint16_t* __restrict__ XAp,
                      const uint16_t* __restrict__ WxPackF,
                      const uint16_t* __restrict__ WxPackB,
                      const float* __restrict__ biasF,
                      const float* __restrict__ biasB,
                      uint16_t* __restrict__ XZf,
                      uint16_t* __restrict__ XZb) {
  const int dir = blockIdx.z;
  const uint16_t* Wp  = dir ? WxPackB : WxPackF;
  const float*   bias = dir ? biasB : biasF;
  uint16_t*      XZ   = dir ? XZb : XZf;

  const int w    = threadIdx.x >> 5;
  const int lane = threadIdx.x & 31;
  const int wm   = w & 3, wn = w >> 2;          // 4 x 2 wave grid
  const int lrow = lane & 15;

  const int mtBase = blockIdx.x * 8 + wm * 2;   // m-tile index (16 rows each)
  const int ntBase = blockIdx.y * 8 + wn * 4;   // n-tile index (16 cols each)

  v8f acc[2][4];
#pragma unroll
  for (int mi = 0; mi < 2; ++mi)
#pragma unroll
    for (int ni = 0; ni < 4; ++ni)
#pragma unroll
      for (int r = 0; r < 8; ++r) acc[mi][ni][r] = 0.0f;

#pragma unroll 4
  for (int kt = 0; kt < KTX_; ++kt) {
    // ---- batch ALL loads for this k-step ----
    ABFrag a[2];
#pragma unroll
    for (int mi = 0; mi < 2; ++mi) {
      const uint16_t* ap =
          XAp + ((size_t)(mtBase + mi) * KTX_ + kt) * 512 + lane * 16;
      a[mi].q[0] = *(const v4u*)(ap);
      a[mi].q[1] = *(const v4u*)(ap + 8);
    }
    ABFrag bfr[4];
#pragma unroll
    for (int ni = 0; ni < 4; ++ni) {
      const uint16_t* bp =
          Wp + ((size_t)kt * NT_ + (ntBase + ni)) * 512 + lane * 16;
      bfr[ni].q[0] = *(const v4u*)(bp);
      bfr[ni].q[1] = *(const v4u*)(bp + 8);
    }
    if (kt + 1 < KTX_)  // hint next B tile toward the WGP (global_prefetch_b8)
      __builtin_prefetch(
          Wp + ((size_t)((kt + 1) * NT_ + ntBase)) * 512 + lane * 16, 0, 3);
    // ---- WMMA chain ----
#pragma unroll
    for (int ni = 0; ni < 4; ++ni)
#pragma unroll
      for (int mi = 0; mi < 2; ++mi)
        acc[mi][ni] = __builtin_amdgcn_wmma_f32_16x16x32_bf16(
            false, a[mi].v, false, bfr[ni].v, (short)0, acc[mi][ni], false, false);
  }

  // Epilogue: add bias, store bf16 in C-fragment packed order (b128/lane).
#pragma unroll
  for (int mi = 0; mi < 2; ++mi) {
    int mt = mtBase + mi;
#pragma unroll
    for (int ni = 0; ni < 4; ++ni) {
      int nt = ntBase + ni;
      float bb = bias[nt * 16 + lrow];
      CQ o;
#pragma unroll
      for (int r = 0; r < 8; ++r) o.u[r] = f2bf_u(acc[mi][ni][r] + bb);
      *(v4u*)(XZ + ((size_t)mt * NT_ + nt) * 256 + lane * 8) = o.q;
    }
  }
}

// ---------------------------------------------------------------------------
// LSTM scan: grid (16 batch-chunks, 2 dirs), 256 threads (8 waves).
// Each step: z = xz_t + h @ Wh  (M=16, N=1024, K=256) via bf16 WMMA, with
// h (bf16) in LDS as the A operand.  Wave w owns hidden cols [w*32, w*32+32)
// of ALL four gates, so gate math stays in registers; c stays in registers.
// Per k-step, the A fragment and all 8 Wh fragments are loaded before the
// 8-WMMA chain.
// ---------------------------------------------------------------------------
__global__ __launch_bounds__(256)
void lstm_scan_kernel(const uint16_t* __restrict__ XZf,
                      const uint16_t* __restrict__ XZb,
                      const uint16_t* __restrict__ WhPackF,
                      const uint16_t* __restrict__ WhPackB,
                      float* __restrict__ hFinF,
                      float* __restrict__ hFinB) {
  __shared__ uint16_t hbuf[16 * 264];   // 16 rows x 256 cols, stride 264 (pad)

  const int bc  = blockIdx.x;
  const int dir = blockIdx.y;
  const uint16_t* XZ = dir ? XZb : XZf;
  const uint16_t* Wp = dir ? WhPackB : WhPackF;
  float* hFin = dir ? hFinB : hFinF;

  const int w    = threadIdx.x >> 5;
  const int lane = threadIdx.x & 31;
  const int lrow = lane & 15;
  const int lhi  = lane >> 4;
  const int klo  = lhi * 8;

  for (int i = threadIdx.x; i < 16 * 264; i += 256) hbuf[i] = 0;
  float cReg[2][8];
#pragma unroll
  for (int cb = 0; cb < 2; ++cb)
#pragma unroll
    for (int r = 0; r < 8; ++r) cReg[cb][r] = 0.0f;
  __syncthreads();

  for (int s = 0; s < T_; ++s) {
    const int t  = dir ? (T_ - 1 - s) : s;
    const int mt = t * 16 + bc;

    // acc <- xz tile (C-fragment packed bf16, one b128 load per tile)
    v8f acc[4][2];
#pragma unroll
    for (int g = 0; g < 4; ++g)
#pragma unroll
      for (int cb = 0; cb < 2; ++cb) {
        int nt = g * 16 + w * 2 + cb;
        CQ x;
        x.q = *(const v4u*)(XZ + ((size_t)mt * NT_ + nt) * 256 + lane * 8);
        v8f a;
#pragma unroll
        for (int r = 0; r < 8; ++r) a[r] = bf2f(x.u[r]);
        acc[g][cb] = a;
      }

    // K loop over h (256) in steps of 32
#pragma unroll
    for (int kt = 0; kt < KTH_; ++kt) {
      // ---- batch loads: A from LDS, 8 Wh fragments from L2 ----
      ABFrag af;
      const uint16_t* hp = hbuf + lrow * 264 + kt * 32 + klo;
      af.q[0] = *(const v4u*)(hp);
      af.q[1] = *(const v4u*)(hp + 16);
      ABFrag bfr[8];
#pragma unroll
      for (int g = 0; g < 4; ++g)
#pragma unroll
        for (int cb = 0; cb < 2; ++cb) {
          int nt = g * 16 + w * 2 + cb;
          const uint16_t* bp = Wp + ((size_t)kt * NT_ + nt) * 512 + lane * 16;
          bfr[g * 2 + cb].q[0] = *(const v4u*)(bp);
          bfr[g * 2 + cb].q[1] = *(const v4u*)(bp + 8);
        }
      // ---- WMMA chain ----
#pragma unroll
      for (int g = 0; g < 4; ++g)
#pragma unroll
        for (int cb = 0; cb < 2; ++cb)
          acc[g][cb] = __builtin_amdgcn_wmma_f32_16x16x32_bf16(
              false, af.v, false, bfr[g * 2 + cb].v, (short)0, acc[g][cb],
              false, false);
    }

    // Gates (Keras order i,f,g,o) — all in registers.
    float hval[2][8];
#pragma unroll
    for (int cb = 0; cb < 2; ++cb)
#pragma unroll
      for (int r = 0; r < 8; ++r) {
        float iv = sigmoidf(acc[0][cb][r]);
        float fv = sigmoidf(acc[1][cb][r]);
        float gv = tanhf(acc[2][cb][r]);
        float ov = sigmoidf(acc[3][cb][r]);
        float c  = fv * cReg[cb][r] + iv * gv;
        cReg[cb][r] = c;
        hval[cb][r] = ov * tanhf(c);
      }

    __syncthreads();   // all A-reads of old h are done
#pragma unroll
    for (int cb = 0; cb < 2; ++cb)
#pragma unroll
      for (int r = 0; r < 8; ++r) {
        int m = r + lhi * 8;
        int u = w * 32 + cb * 16 + lrow;
        hbuf[m * 264 + u] = f2bf_u(hval[cb][r]);
        if (s == T_ - 1)
          hFin[(size_t)(bc * 16 + m) * 256 + u] = hval[cb][r];
      }
    __syncthreads();   // new h visible before next step's A-reads
  }
}

// ---------------------------------------------------------------------------
// MLP head (tiny; fp32 VALU). Feature quirk of the reference: cols 0..511 are
// row-0 of concat(h_f, h_b); cols 512..524 are stats[0, 9:22]; cols 525..533
// are stats[i, 0:9].
// ---------------------------------------------------------------------------
__global__ void mlp1_kernel(const float* __restrict__ hF,
                            const float* __restrict__ hB,
                            const float* __restrict__ stats,
                            const float* __restrict__ W1,
                            const float* __restrict__ b1,
                            float* __restrict__ H1) {
  int idx = blockIdx.x * blockDim.x + threadIdx.x;
  if (idx >= B_ * H1_) return;
  int i = idx / H1_, j = idx % H1_;
  float acc = b1[j];
  for (int k = 0; k < FEAT_; ++k) {
    float fv;
    if (k < 256)      fv = hF[k];                     // h_f[0][k]
    else if (k < 512) fv = hB[k - 256];               // h_b[0][k-256]
    else if (k < 525) fv = stats[9 + (k - 512)];      // stats[0, 9:22]
    else              fv = stats[i * 22 + (k - 525)]; // stats[i, 0:9]
    acc += fv * W1[(size_t)k * H1_ + j];
  }
  H1[idx] = fmaxf(acc, 0.0f);
}

__global__ void mlp2_kernel(const float* __restrict__ H1,
                            const float* __restrict__ W2,
                            const float* __restrict__ b2,
                            float* __restrict__ H2) {
  int idx = blockIdx.x * blockDim.x + threadIdx.x;
  if (idx >= B_ * H2_) return;
  int i = idx / H2_, j = idx % H2_;
  float acc = b2[j];
  for (int k = 0; k < H1_; ++k)
    acc += H1[(size_t)i * H1_ + k] * W2[(size_t)k * H2_ + j];
  H2[idx] = fmaxf(acc, 0.0f);
}

__global__ void mlp3_kernel(const float* __restrict__ H2,
                            const float* __restrict__ Wp,
                            const float* __restrict__ bp,
                            float* __restrict__ out) {
  int i = blockIdx.x * blockDim.x + threadIdx.x;
  if (i >= B_) return;
  float l[3];
#pragma unroll
  for (int c = 0; c < 3; ++c) {
    float acc = bp[c];
    for (int k = 0; k < H2_; ++k) acc += H2[(size_t)i * H2_ + k] * Wp[k * 3 + c];
    l[c] = acc;
  }
  float mx = fmaxf(l[0], fmaxf(l[1], l[2]));
  float e0 = __expf(l[0] - mx), e1 = __expf(l[1] - mx), e2 = __expf(l[2] - mx);
  float s = e0 + e1 + e2;
  out[i * 3 + 0] = e0 / s;
  out[i * 3 + 1] = e1 / s;
  out[i * 3 + 2] = e2 / s;
}

// ---------------------------------------------------------------------------
extern "C" void kernel_launch(void* const* d_in, const int* in_sizes, int n_in,
                              void* d_out, int out_size, void* d_ws, size_t ws_size,
                              hipStream_t stream) {
  const float* X     = (const float*)d_in[0];
  const float* stats = (const float*)d_in[1];
  const float* Wx_f  = (const float*)d_in[2];
  const float* Wh_f  = (const float*)d_in[3];
  const float* b_f   = (const float*)d_in[4];
  const float* Wx_b  = (const float*)d_in[5];
  const float* Wh_b  = (const float*)d_in[6];
  const float* b_b   = (const float*)d_in[7];
  const float* W1    = (const float*)d_in[8];
  const float* b1    = (const float*)d_in[9];
  const float* W2    = (const float*)d_in[10];
  const float* b2    = (const float*)d_in[11];
  const float* Wp    = (const float*)d_in[12];
  const float* bp    = (const float*)d_in[13];
  float* out = (float*)d_out;

  char* ws = (char*)d_ws;
  size_t off = 0;
  auto alloc = [&](size_t bytes) -> void* {
    void* p = ws + off;
    off += (bytes + 255) & ~(size_t)255;
    return p;
  };

  const size_t xaPackElems = (size_t)MT_ * KTX_ * 512;  // 98,304,000
  const size_t wxPackElems = (size_t)KTX_ * NT_ * 512;  // 786,432
  const size_t whPackElems = (size_t)KTH_ * NT_ * 512;  // 262,144
  const size_t xzElems     = (size_t)MT_ * NT_ * 256;   // 131,072,000

  uint16_t* XAp  = (uint16_t*)alloc(xaPackElems * 2);
  uint16_t* WxFp = (uint16_t*)alloc(wxPackElems * 2);
  uint16_t* WxBp = (uint16_t*)alloc(wxPackElems * 2);
  uint16_t* WhFp = (uint16_t*)alloc(whPackElems * 2);
  uint16_t* WhBp = (uint16_t*)alloc(whPackElems * 2);
  uint16_t* XZf  = (uint16_t*)alloc(xzElems * 2);
  uint16_t* XZb  = (uint16_t*)alloc(xzElems * 2);
  float* hFinF   = (float*)alloc((size_t)B_ * U_ * 4);
  float* hFinB   = (float*)alloc((size_t)B_ * U_ * 4);
  float* H1b     = (float*)alloc((size_t)B_ * H1_ * 4);
  float* H2b     = (float*)alloc((size_t)B_ * H2_ * 4);

  // 1) Pack activations and weights into WMMA fragment order (bf16).
  pack_a_kernel<<<((int)xaPackElems + 255) / 256, 256, 0, stream>>>(X, XAp, (int)xaPackElems);
  pack_b_kernel<<<((int)wxPackElems + 255) / 256, 256, 0, stream>>>(Wx_f, WxFp, (int)wxPackElems, G_);
  pack_b_kernel<<<((int)wxPackElems + 255) / 256, 256, 0, stream>>>(Wx_b, WxBp, (int)wxPackElems, G_);
  pack_b_kernel<<<((int)whPackElems + 255) / 256, 256, 0, stream>>>(Wh_f, WhFp, (int)whPackElems, G_);
  pack_b_kernel<<<((int)whPackElems + 255) / 256, 256, 0, stream>>>(Wh_b, WhBp, (int)whPackElems, G_);

  // 2) Input projections (both directions), WMMA bf16.
  proj_gemm_kernel<<<dim3(1000, 8, 2), 256, 0, stream>>>(
      XAp, WxFp, WxBp, b_f, b_b, XZf, XZb);

  // 3) Serial LSTM scans (both directions), WMMA bf16 recurrence.
  lstm_scan_kernel<<<dim3(16, 2), 256, 0, stream>>>(
      XZf, XZb, WhFp, WhBp, hFinF, hFinB);

  // 4) MLP head + softmax.
  mlp1_kernel<<<(B_ * H1_ + 255) / 256, 256, 0, stream>>>(hFinF, hFinB, stats, W1, b1, H1b);
  mlp2_kernel<<<(B_ * H2_ + 255) / 256, 256, 0, stream>>>(H1b, W2, b2, H2b);
  mlp3_kernel<<<1, 256, 0, stream>>>(H2b, Wp, bp, out);
}
